// HierarchicalQuantizer_88708254531960
// MI455X (gfx1250) — matrix operand order repaired and checked by reading, not compiled, not run on hardware
//
#include <hip/hip_runtime.h>
#include <hip/hip_bf16.h>

typedef __attribute__((ext_vector_type(16))) __bf16 v16bf;
typedef __attribute__((ext_vector_type(8)))  float  v8f;

#define D_DIM   256
#define K_CODES 4096
#define BN_TOK  65536
#define COMMIT  0.25f

// ---------------- prep: bf16 codebook + ||e||^2 per code ----------------
__global__ void __launch_bounds__(256) vq_prep(const float* __restrict__ cb,
                                               __bf16* __restrict__ cbh,
                                               float* __restrict__ enorm) {
    const int wave = threadIdx.x >> 5;
    const int lane = threadIdx.x & 31;
    const int code = blockIdx.x * 8 + wave;            // one code per wave
    const float* src = cb + (size_t)code * D_DIM + lane * 8;

    float v[8];
    *(float4*)&v[0] = *(const float4*)&src[0];
    *(float4*)&v[4] = *(const float4*)&src[4];

    __bf16 h[8];
    float ss = 0.0f;
#pragma unroll
    for (int i = 0; i < 8; ++i) { ss += v[i] * v[i]; h[i] = (__bf16)v[i]; }

    *(float4*)&cbh[(size_t)code * D_DIM + lane * 8] = *(float4*)&h[0];

#pragma unroll
    for (int o = 16; o > 0; o >>= 1) ss += __shfl_xor(ss, o, 32);
    if (lane == 0) enorm[code] = ss;
}

// ---------------- zero scratch (counts + loss accumulator) ----------------
__global__ void vq_zero(int* __restrict__ counts, float* __restrict__ loss) {
    int i = blockIdx.x * 256 + threadIdx.x;
    if (i < K_CODES) counts[i] = 0;
    if (i == 0) loss[0] = 0.0f;
}

// async global -> LDS: move 32 contiguous bytes per lane (2 x b128), ASYNCcnt path
__device__ __forceinline__ void async_tile_load32(const __bf16* gsrc, void* ldst) {
    unsigned           lds = (unsigned)(uintptr_t)ldst;           // LDS byte offset
    unsigned long long ga  = (unsigned long long)(uintptr_t)gsrc; // global address
    asm volatile(
        "global_load_async_to_lds_b128 %0, %1, off\n\t"
        "global_load_async_to_lds_b128 %0, %1, off offset:16"
        :: "v"(lds), "v"(ga) : "memory");
}

// ---------------- fused distance GEMM + argmin ----------------
// grid = BN/128 blocks, 256 threads (8 waves); each wave owns 16 token rows.
__global__ void __launch_bounds__(256) vq_argmin(const float* __restrict__ x,
                                                 const __bf16* __restrict__ cbh,
                                                 const float* __restrict__ enorm,
                                                 float* __restrict__ idx_out) {
    __shared__ __align__(32) __bf16 sB[3][16 * D_DIM];  // 3 x 8 KB codebook tiles
    __shared__ float sXn[8][16];

    const int tid  = threadIdx.x;
    const int wave = tid >> 5;
    const int lane = tid & 31;
    const int col  = lane & 15;
    const int half = lane >> 4;
    const int m0   = blockIdx.x * 128 + wave * 16;
    const int row  = m0 + col;                          // this lane's A-row
    constexpr int NT = K_CODES / 16;                    // 256 code tiles

    // kick off async staging of tiles 0 and 1 (2 x b128 per thread per tile)
    async_tile_load32(cbh + 0 * 16 * D_DIM + tid * 16, &sB[0][tid * 16]);
    async_tile_load32(cbh + 1 * 16 * D_DIM + tid * 16, &sB[1][tid * 16]);

    // ---- A tile: 16 tokens x 256 dims -> bf16 regs (contiguous k order: element s = k 16*half+s) ----
    // lanes 0-15 hold K = 32k+[0..15]; lanes 16-31 hold K = 32k+[16..31]
    const float* xr = x + (size_t)row * D_DIM;
    v16bf a[8];
    float ss = 0.0f;
#pragma unroll
    for (int kk = 0; kk < 8; ++kk) {
        const int base = kk * 32 + half * 16;
        float lo[8], hi[8];
        *(float4*)&lo[0] = *(const float4*)&xr[base];
        *(float4*)&lo[4] = *(const float4*)&xr[base + 4];
        *(float4*)&hi[0] = *(const float4*)&xr[base + 8];
        *(float4*)&hi[4] = *(const float4*)&xr[base + 12];
#pragma unroll
        for (int i = 0; i < 8; ++i) {
            ss += lo[i] * lo[i] + hi[i] * hi[i];
            a[kk][i]     = (__bf16)lo[i];
            a[kk][8 + i] = (__bf16)hi[i];
        }
    }
    ss += __shfl_xor(ss, 16, 32);                      // both K-halves of the row
    if (half == 0) sXn[wave][col] = ss;
    __syncthreads();

    float xn[8];
#pragma unroll
    for (int j = 0; j < 8; ++j) xn[j] = sXn[wave][j + 8 * half];

    float best[8];
    int   bidx[8];
#pragma unroll
    for (int j = 0; j < 8; ++j) { best[j] = 3.4e38f; bidx[j] = 0; }

    float en_cur = enorm[col];                          // ||e||^2, tile 0

    for (int t = 0; t < NT; ++t) {
        // tile t's data complete once our two oldest async loads retire
        if (t + 1 < NT) asm volatile("s_wait_asynccnt 0x2" ::: "memory");
        else            asm volatile("s_wait_asynccnt 0x0" ::: "memory");
        __syncthreads();    // all waves' portions of tile t landed; buf (t+2)%3 free

        if (t + 2 < NT)
            async_tile_load32(cbh + (size_t)(t + 2) * 16 * D_DIM + tid * 16,
                              &sB[(t + 2) % 3][tid * 16]);
        const float en_nxt = (t + 1 < NT) ? enorm[(t + 1) * 16 + col] : 0.0f;

        const __bf16* bb = &sB[t % 3][0];
        v8f c0 = {};
        v8f c1 = {};
#pragma unroll
        for (int kk = 0; kk < 8; kk += 2) {
            // B layout: lane = column (code); lanes 0-15 K=[0..15], 16-31 K=[16..31]
            const v16bf b0 = *(const v16bf*)&bb[col * D_DIM + kk * 32 + half * 16];
            const v16bf b1 = *(const v16bf*)&bb[col * D_DIM + (kk + 1) * 32 + half * 16];
            c0 = __builtin_amdgcn_wmma_f32_16x16x32_bf16(false, a[kk],     false, b0,
                                                         (short)0, c0, false, false);
            c1 = __builtin_amdgcn_wmma_f32_16x16x32_bf16(false, a[kk + 1], false, b1,
                                                         (short)0, c1, false, false);
        }

        const int code = t * 16 + col;
#pragma unroll
        for (int j = 0; j < 8; ++j) {
            float dsq = xn[j] + en_cur - 2.0f * (c0[j] + c1[j]);
            dsq = fmaxf(dsq, 0.0f);
            if (dsq < best[j]) { best[j] = dsq; bidx[j] = code; }
        }
        en_cur = en_nxt;
    }

    // cross-lane argmin over the 16 candidate columns (tie -> lowest index)
#pragma unroll
    for (int j = 0; j < 8; ++j) {
        float bv = best[j];
        int   bi = bidx[j];
#pragma unroll
        for (int o = 8; o > 0; o >>= 1) {
            float ov = __shfl_xor(bv, o, 16);
            int   oi = __shfl_xor(bi, o, 16);
            if (ov < bv || (ov == bv && oi < bi)) { bv = ov; bi = oi; }
        }
        if (col == 0) idx_out[m0 + 8 * half + j] = (float)bi;
    }
}

// ---------------- gather q = codebook[idx] + fused loss/histogram ----------------
__global__ void __launch_bounds__(256) vq_gather(const float* __restrict__ x,
                                                 const float* __restrict__ cb,
                                                 const float* __restrict__ idx_f,
                                                 float* __restrict__ q_out,
                                                 float* __restrict__ loss_acc,
                                                 int* __restrict__ counts) {
    __shared__ float red[256];
    const int tok  = blockIdx.x;
    const int d    = threadIdx.x;
    const int code = (int)idx_f[tok];

    float q  = cb[(size_t)code * D_DIM + d];
    float xv = x[(size_t)tok * D_DIM + d];
    q_out[(size_t)tok * D_DIM + d] = q;                // q_st == q in forward value

    float diff = q - xv;
    red[d] = diff * diff;
    __syncthreads();
#pragma unroll
    for (int s = 128; s > 0; s >>= 1) {
        if (d < s) red[d] += red[d + s];
        __syncthreads();
    }
    if (d == 0) {
        atomicAdd(loss_acc, red[0]);
        atomicAdd(&counts[code], 1);
    }
}

// ---------------- finalize: loss scalar + perplexity ----------------
__global__ void __launch_bounds__(256) vq_finalize(const int* __restrict__ counts,
                                                   const float* __restrict__ loss_acc,
                                                   float* __restrict__ out_loss,
                                                   float* __restrict__ out_perp) {
    __shared__ float red[256];
    float acc = 0.0f;
    for (int k = threadIdx.x; k < K_CODES; k += 256) {
        float p = (float)counts[k] * (1.0f / (float)BN_TOK);
        acc += p * logf(p + 1e-10f);
    }
    red[threadIdx.x] = acc;
    __syncthreads();
#pragma unroll
    for (int s = 128; s > 0; s >>= 1) {
        if (threadIdx.x < s) red[threadIdx.x] += red[threadIdx.x + s];
        __syncthreads();
    }
    if (threadIdx.x == 0) {
        out_perp[0] = expf(-red[0]);
        // codebook_loss + COMMIT*commit_loss: forward values identical -> 1.25x
        out_loss[0] = (1.0f + COMMIT) * loss_acc[0] / ((float)BN_TOK * (float)D_DIM);
    }
}

extern "C" void kernel_launch(void* const* d_in, const int* in_sizes, int n_in,
                              void* d_out, int out_size, void* d_ws, size_t ws_size,
                              hipStream_t stream) {
    const float* x  = (const float*)d_in[0];   // [16,4096,256] f32
    const float* cb = (const float*)d_in[1];   // [4096,256]    f32

    float* out      = (float*)d_out;
    float* q_out    = out;                                  // 16,777,216
    float* out_loss = out + (size_t)BN_TOK * D_DIM;         // +1
    float* out_perp = out_loss + 1;                         // +1
    float* idx_out  = out_perp + 1;                         // +65,536

    char*    ws       = (char*)d_ws;
    __bf16*  cbh      = (__bf16*)ws;                                          // 2 MB
    float*   enorm    = (float*)(ws + (size_t)K_CODES * D_DIM * 2);           // 16 KB
    int*     counts   = (int*)  (ws + (size_t)K_CODES * D_DIM * 2 + K_CODES * 4);
    float*   loss_acc = (float*)(ws + (size_t)K_CODES * D_DIM * 2 + K_CODES * 8);

    vq_zero   <<<(K_CODES + 255) / 256, 256, 0, stream>>>(counts, loss_acc);
    vq_prep   <<<K_CODES / 8,           256, 0, stream>>>(cb, cbh, enorm);
    vq_argmin <<<BN_TOK / 128,          256, 0, stream>>>(x, cbh, enorm, idx_out);
    vq_gather <<<BN_TOK,                256, 0, stream>>>(x, cb, idx_out, q_out,
                                                          loss_acc, counts);
    vq_finalize<<<1,                    256, 0, stream>>>(counts, loss_acc,
                                                          out_loss, out_perp);
}
